// Attention2_80169859547543
// MI455X (gfx1250) — matrix operand order repaired and checked by reading, not compiled, not run on hardware
//
#include <hip/hip_runtime.h>
#include <hip/hip_bf16.h>

// ---------------------------------------------------------------------------
// Problem constants (from reference)
// ---------------------------------------------------------------------------
#define ENC_DIM 1024
#define DEC_DIM 512
#define HID     1024
#define HEADS   8
#define BATCH   4096

typedef __bf16 bf16_t;
typedef __attribute__((ext_vector_type(16))) __bf16 v16bf;
typedef __attribute__((ext_vector_type(8)))  __bf16 v8bf;
typedef __attribute__((ext_vector_type(4)))  __bf16 v4bf;
typedef __attribute__((ext_vector_type(8)))  float  v8f;

static __device__ __forceinline__ bf16_t f2bf(float f) { return (bf16_t)f; }

// ---------------------------------------------------------------------------
// Tiled WMMA GEMM:  D = act(A[M,K] * Bw[K,N] + bias[N])
//   block = 256 threads = 8 waves; tile BM=256 x BN=64 x BK=32
//   wave w owns rows [w*32, w*32+32) x all 64 cols -> 8 wmma accumulators
//   register-staged software pipeline: global loads for step k+1 are issued
//   before the WMMAs of step k, and committed to LDS after the tail barrier.
//   blockIdx.z selects a weight/bias/output slice (for the 8 heads)
// ---------------------------------------------------------------------------
constexpr int BM = 256, BN = 64, BK = 32;
constexpr int ASTR = BK + 8;   // 40 halves (80 B row stride: 16B-aligned, conflict-free)
constexpr int BSTR = BK + 8;   // 40 halves (B stored transposed: [n][k])

template <typename AT, typename OT, bool RELU>
__global__ __launch_bounds__(256) void gemm_bias_act(
    const AT* __restrict__ A, const float* __restrict__ Bw,
    const float* __restrict__ bias, OT* __restrict__ D,
    int M, int N, int K,
    long wstride, long bstride, long dstride)
{
  __shared__ bf16_t As[BM * ASTR];   // 20.0 KB
  __shared__ bf16_t Bs[BN * BSTR];   //  5.0 KB

  const int z = blockIdx.z;
  const float* Wz = Bw   + (long)z * wstride;
  const float* bz = bias + (long)z * bstride;
  OT*          Dz = D    + (long)z * dstride;

  const int n0   = blockIdx.x * BN;
  const int m0   = blockIdx.y * BM;
  const int tid  = threadIdx.x;
  const int wave = tid >> 5;
  const int lane = tid & 31;
  const int lh   = lane >> 4;    // 0/1: which K-half this lane holds
  const int l15  = lane & 15;

  v8f acc[2][4] = {};            // 2 row-subtiles x 4 col-subtiles (16x16 f32)

  // A tile global<->LDS mapping: 8 passes, 32 rows/pass, 8 lanes per row
  const int a_r = tid >> 3;            // row-within-pass 0..31
  const int a_c = (tid & 7) * 4;       // col 0..28 (4 elems)
  // B tile mapping: 8 threads per K-row, 8 cols each
  const int b_r = tid >> 3;            // k row 0..31
  const int b_c = (tid & 7) * 8;       // n col 0..56

  // ---- register staging for the software pipeline ------------------------
  v4bf   aR[8];
  bf16_t bR[8];

  auto g_load = [&](int kk) {
    // A tile (BM x BK): convert to bf16 in registers
#pragma unroll
    for (int p = 0; p < 8; ++p) {
      const int row = p * 32 + a_r;
      if constexpr (sizeof(AT) == 4) {
        const float4 x = *(const float4*)(A + (long)(m0 + row) * K + kk + a_c);
        v4bf h;
        h[0] = f2bf(x.x); h[1] = f2bf(x.y); h[2] = f2bf(x.z); h[3] = f2bf(x.w);
        aR[p] = h;
      } else {
        aR[p] = *(const v4bf*)(A + (long)(m0 + row) * K + kk + a_c);
      }
    }
    // B tile (BK x BN)
    const float4* src = (const float4*)(Wz + (long)(kk + b_r) * N + n0 + b_c);
    const float4 y0 = src[0], y1 = src[1];
    bR[0] = f2bf(y0.x); bR[1] = f2bf(y0.y); bR[2] = f2bf(y0.z); bR[3] = f2bf(y0.w);
    bR[4] = f2bf(y1.x); bR[5] = f2bf(y1.y); bR[6] = f2bf(y1.z); bR[7] = f2bf(y1.w);
  };

  auto s_commit = [&]() {
#pragma unroll
    for (int p = 0; p < 8; ++p)
      *(v4bf*)(&As[(p * 32 + a_r) * ASTR + a_c]) = aR[p];
#pragma unroll
    for (int j = 0; j < 8; ++j)
      Bs[(b_c + j) * BSTR + b_r] = bR[j];     // transposed store
  };

  // ---- prologue: stage tile 0 --------------------------------------------
  g_load(0);
  s_commit();

  for (int k0 = 0; k0 < K; k0 += BK) {
    __syncthreads();                       // LDS tile for step k0 visible
    const bool more = (k0 + BK) < K;
    if (more) g_load(k0 + BK);             // prefetch next tile into regs

    // ---- fragment loads (all up front -> one dscnt wait, 8 wmmas b2b) ----
    // A frag (16x32 bf16, ISA layout): lane<16 row=l15 K={0..7,16..23};
    //                                  lane>=16 K={8..15,24..31}
    const int arow = wave * 32 + l15;
    v8bf a0lo = *(const v8bf*)(&As[arow * ASTR + lh * 8]);
    v8bf a0hi = *(const v8bf*)(&As[arow * ASTR + lh * 8 + 16]);
    v8bf a1lo = *(const v8bf*)(&As[(arow + 16) * ASTR + lh * 8]);
    v8bf a1hi = *(const v8bf*)(&As[(arow + 16) * ASTR + lh * 8 + 16]);
    v16bf af0 = __builtin_shufflevector(
        a0lo, a0hi, 0, 1, 2, 3, 4, 5, 6, 7, 8, 9, 10, 11, 12, 13, 14, 15);
    v16bf af1 = __builtin_shufflevector(
        a1lo, a1hi, 0, 1, 2, 3, 4, 5, 6, 7, 8, 9, 10, 11, 12, 13, 14, 15);

    v16bf bf[4];
#pragma unroll
    for (int ct = 0; ct < 4; ++ct) {
      // B frag (32x16): lane<16 col=l15 K={0..7,16..23}; lane>=16 K={8..15,24..31}
      const int bcol = ct * 16 + l15;
      v8bf blo = *(const v8bf*)(&Bs[bcol * BSTR + lh * 8]);
      v8bf bhi = *(const v8bf*)(&Bs[bcol * BSTR + lh * 8 + 16]);
      bf[ct] = __builtin_shufflevector(
          blo, bhi, 0, 1, 2, 3, 4, 5, 6, 7, 8, 9, 10, 11, 12, 13, 14, 15);
    }

#pragma unroll
    for (int ct = 0; ct < 4; ++ct) {
      acc[0][ct] = __builtin_amdgcn_wmma_f32_16x16x32_bf16(
          false, af0, false, bf[ct], (short)0, acc[0][ct], false, false);
      acc[1][ct] = __builtin_amdgcn_wmma_f32_16x16x32_bf16(
          false, af1, false, bf[ct], (short)0, acc[1][ct], false, false);
    }

    __syncthreads();                       // compute done before overwrite
    if (more) s_commit();                  // commit prefetched tile to LDS
  }

  // ---- epilogue: bias (+ReLU), store --------------------------------------
#pragma unroll
  for (int r = 0; r < 2; ++r) {
#pragma unroll
    for (int ct = 0; ct < 4; ++ct) {
      const int col = n0 + ct * 16 + l15;
      const float bv = bz[col];
#pragma unroll
      for (int i = 0; i < 8; ++i) {
        const int row = m0 + wave * 32 + r * 16 + lh * 8 + i;  // C/D lane layout
        float v = acc[r][ct][i] + bv;
        if constexpr (RELU) v = fmaxf(v, 0.0f);
        Dz[(long)row * N + col] = (OT)v;
      }
    }
  }
}

// ---------------------------------------------------------------------------
// Fused scores -> softmax(heads) -> weighted context, one block per batch row.
// heads kept in registers (read exactly once): 8 heads x 4 dims per thread.
// ---------------------------------------------------------------------------
__global__ __launch_bounds__(256) void attn_fuse(
    const bf16_t* __restrict__ heads,   // [HEADS][BATCH][HID] bf16
    const float*  __restrict__ decH,    // [BATCH][HID] f32
    float* __restrict__ out)            // [BATCH][HID] f32
{
  const int b    = blockIdx.x;
  const int tid  = threadIdx.x;
  const int lane = tid & 31;
  const int wave = tid >> 5;
  const int d    = tid * 4;            // HID/256 = 4 dims per thread

  __shared__ float wred[8][HEADS];

  const float4 dh = *(const float4*)(decH + (long)b * HID + d);

  float hv[HEADS][4];
  float s[HEADS];
#pragma unroll
  for (int k = 0; k < HEADS; ++k) {
    const v4bf h = *(const v4bf*)(heads + ((long)k * BATCH + b) * HID + d);
    hv[k][0] = (float)h[0]; hv[k][1] = (float)h[1];
    hv[k][2] = (float)h[2]; hv[k][3] = (float)h[3];
    s[k] = hv[k][0] * dh.x + hv[k][1] * dh.y + hv[k][2] * dh.z + hv[k][3] * dh.w;
  }

  // wave32 butterfly reduction of the 8 partial scores
#pragma unroll
  for (int off = 16; off > 0; off >>= 1)
#pragma unroll
    for (int k = 0; k < HEADS; ++k)
      s[k] += __shfl_xor(s[k], off, 32);

  if (lane == 0)
#pragma unroll
    for (int k = 0; k < HEADS; ++k) wred[wave][k] = s[k];
  __syncthreads();

  float sc[HEADS];
#pragma unroll
  for (int k = 0; k < HEADS; ++k) {
    float t = 0.f;
#pragma unroll
    for (int w = 0; w < 8; ++w) t += wred[w][k];
    sc[k] = t;
  }

  // softmax over 8 heads (computed redundantly per thread; trivial cost)
  float m = sc[0];
#pragma unroll
  for (int k = 1; k < HEADS; ++k) m = fmaxf(m, sc[k]);
  float den = 0.f;
#pragma unroll
  for (int k = 0; k < HEADS; ++k) { sc[k] = __expf(sc[k] - m); den += sc[k]; }
  const float inv = 1.0f / den;

  float4 o = {0.f, 0.f, 0.f, 0.f};
#pragma unroll
  for (int k = 0; k < HEADS; ++k) {
    const float a = sc[k];
    o.x += a * hv[k][0]; o.y += a * hv[k][1];
    o.z += a * hv[k][2]; o.w += a * hv[k][3];
  }
  o.x *= inv; o.y *= inv; o.z *= inv; o.w *= inv;
  *(float4*)(out + (long)b * HID + d) = o;
}

// ---------------------------------------------------------------------------
// Launch
// ---------------------------------------------------------------------------
extern "C" void kernel_launch(void* const* d_in, const int* in_sizes, int n_in,
                              void* d_out, int out_size, void* d_ws, size_t ws_size,
                              hipStream_t stream) {
  const float* enc_in  = (const float*)d_in[0];  // [B, ENC_DIM]
  const float* dec_in  = (const float*)d_in[1];  // [B, DEC_DIM]
  const float* W_enc   = (const float*)d_in[2];  // [ENC_DIM, HID]
  const float* b_enc   = (const float*)d_in[3];  // [HID]
  const float* W_heads = (const float*)d_in[4];  // [HEADS, HID, HID]
  const float* b_heads = (const float*)d_in[5];  // [HEADS, HID]
  const float* W_dec   = (const float*)d_in[6];  // [DEC_DIM, HID]
  const float* b_dec   = (const float*)d_in[7];  // [HID]
  float* out = (float*)d_out;

  // workspace layout
  char* ws = (char*)d_ws;
  bf16_t* enc_h = (bf16_t*)ws;                                   //  8 MB
  size_t off = (size_t)BATCH * HID * sizeof(bf16_t);
  bf16_t* headsb = (bf16_t*)(ws + off);                          // 64 MB
  off += (size_t)HEADS * BATCH * HID * sizeof(bf16_t);
  float* decH = (float*)(ws + off);                              // 16.8 MB

  dim3 blk(256);

  // 1) enc_h = relu(enc_in @ W_enc + b_enc)          [4096 x 1024, K=1024]
  dim3 g1(HID / BN, BATCH / BM, 1);
  gemm_bias_act<float, bf16_t, true><<<g1, blk, 0, stream>>>(
      enc_in, W_enc, b_enc, enc_h, BATCH, HID, ENC_DIM, 0, 0, 0);

  // 2) heads[k] = relu(enc_h @ W_heads[k] + b_heads[k])  (z = 8 heads)
  dim3 g2(HID / BN, BATCH / BM, HEADS);
  gemm_bias_act<bf16_t, bf16_t, true><<<g2, blk, 0, stream>>>(
      enc_h, W_heads, b_heads, headsb, BATCH, HID, HID,
      (long)HID * HID, (long)HID, (long)BATCH * HID);

  // 3) dec_H = relu(dec_in @ W_dec + b_dec)          [4096 x 1024, K=512]
  dim3 g3(HID / BN, BATCH / BM, 1);
  gemm_bias_act<float, float, true><<<g3, blk, 0, stream>>>(
      dec_in, W_dec, b_dec, decH, BATCH, HID, DEC_DIM, 0, 0, 0);

  // 4) fused scores -> softmax -> context
  attn_fuse<<<dim3(BATCH), blk, 0, stream>>>(headsb, decH, out);
}
